// MultiAttentionWithGating_27565100106134
// MI455X (gfx1250) — compile-verified, matrix operand order checked by reading, not run on hardware
//
#include <hip/hip_runtime.h>
#include <hip/hip_bf16.h>

// Problem constants (match reference)
#define B_  4
#define T_  2048
#define C_  1024
#define NH_ 16
#define HS_ 64

typedef __attribute__((ext_vector_type(16))) __bf16 v16bf;
typedef __attribute__((ext_vector_type(8)))  float  v8f;

#if __has_builtin(__builtin_amdgcn_global_load_async_to_lds_b128) && \
    __has_builtin(__builtin_amdgcn_s_wait_asynccnt)
#define USE_ASYNC_LDS 1
// Exact param types per hipcc diagnostic: v4i32 in AS(1) / AS(3)
typedef int v4i_ __attribute__((vector_size(16)));
typedef __attribute__((address_space(1))) v4i_* gas_v4i;
typedef __attribute__((address_space(3))) v4i_* las_v4i;
#else
#define USE_ASYNC_LDS 0
#endif

union Frag {
  v16bf v;
  uint4 q[2];
};

__device__ __forceinline__ unsigned short f2bf(float f) {
  unsigned int u = __builtin_bit_cast(unsigned int, f);
  u += 0x7FFFu + ((u >> 16) & 1u);          // round-to-nearest-even
  return (unsigned short)(u >> 16);
}

__device__ __forceinline__ v8f wmma_bf16(const Frag& a, const Frag& b, v8f c) {
  // 8 args: (neg_a, A, neg_b, B, c_mod, C, reuse_a, reuse_b)
  return __builtin_amdgcn_wmma_f32_16x16x32_bf16(false, a.v, false, b.v,
                                                 (short)0, c, false, false);
}

// ---------------------------------------------------------------------------
// fp32 -> bf16 conversion (vectorized, grid-stride)
// ---------------------------------------------------------------------------
__global__ void cvt_f32_bf16(const float* __restrict__ in,
                             unsigned short* __restrict__ out, int n4) {
  int i = blockIdx.x * blockDim.x + threadIdx.x;
  int stride = gridDim.x * blockDim.x;
  for (; i < n4; i += stride) {
    float4 f = ((const float4*)in)[i];
    ushort4 o;
    o.x = f2bf(f.x); o.y = f2bf(f.y); o.z = f2bf(f.z); o.w = f2bf(f.w);
    ((ushort4*)out)[i] = o;
  }
}

// ---------------------------------------------------------------------------
// GEMM core: one wave computes a 32(M) x 64(N) tile of A(MxK) * B(NxK)^T.
// Each B fragment feeds two WMMAs (two M sub-tiles) -> ~40% fewer loads/FLOP.
// A/B are row-major bf16 with K contiguous: the 16-bit WMMA A layout per lane
// is K = {kb..kb+7} and {kb+16..kb+23} (kb = (lane>=16)*8) -> two b128 loads.
// ---------------------------------------------------------------------------
__device__ __forceinline__ void gemm_tile_32x64(
    const unsigned short* __restrict__ A, const unsigned short* __restrict__ Bm,
    int lda, int ldb, int m0, int n0, int K, int lane,
    v8f accLo[4], v8f accHi[4]) {
  const int row = lane & 15;
  const int kb  = (lane >> 4) ? 8 : 0;
  const unsigned short* arow0 = A + (size_t)(m0 + row) * lda;
  const unsigned short* arow1 = A + (size_t)(m0 + 16 + row) * lda;
  const unsigned short* bp0 = Bm + (size_t)(n0 +  0 + row) * ldb;
  const unsigned short* bp1 = Bm + (size_t)(n0 + 16 + row) * ldb;
  const unsigned short* bp2 = Bm + (size_t)(n0 + 32 + row) * ldb;
  const unsigned short* bp3 = Bm + (size_t)(n0 + 48 + row) * ldb;
  for (int k = 0; k < K; k += 32) {
    // stream-prefetch the HBM-resident A rows well ahead of use
    __builtin_prefetch(arow0 + k + 512, 0, 1);
    __builtin_prefetch(arow1 + k + 512, 0, 1);
    Frag a0, a1;
    a0.q[0] = *(const uint4*)(arow0 + k + kb);
    a0.q[1] = *(const uint4*)(arow0 + k + kb + 16);
    a1.q[0] = *(const uint4*)(arow1 + k + kb);
    a1.q[1] = *(const uint4*)(arow1 + k + kb + 16);
    Frag b;
    b.q[0] = *(const uint4*)(bp0 + k + kb);
    b.q[1] = *(const uint4*)(bp0 + k + kb + 16);
    accLo[0] = wmma_bf16(a0, b, accLo[0]);
    accHi[0] = wmma_bf16(a1, b, accHi[0]);
    b.q[0] = *(const uint4*)(bp1 + k + kb);
    b.q[1] = *(const uint4*)(bp1 + k + kb + 16);
    accLo[1] = wmma_bf16(a0, b, accLo[1]);
    accHi[1] = wmma_bf16(a1, b, accHi[1]);
    b.q[0] = *(const uint4*)(bp2 + k + kb);
    b.q[1] = *(const uint4*)(bp2 + k + kb + 16);
    accLo[2] = wmma_bf16(a0, b, accLo[2]);
    accHi[2] = wmma_bf16(a1, b, accHi[2]);
    b.q[0] = *(const uint4*)(bp3 + k + kb);
    b.q[1] = *(const uint4*)(bp3 + k + kb + 16);
    accLo[3] = wmma_bf16(a0, b, accLo[3]);
    accHi[3] = wmma_bf16(a1, b, accHi[3]);
  }
}

// ---------------------------------------------------------------------------
// qkv = x @ W_attn^T ; scatter into Q[B,NH,T,HS], K[B,NH,T,HS], Vt[B,NH,HS,T]
// block = 4 waves, each wave a 32x64 tile; block tile = 32(M) x 256(N)
// ---------------------------------------------------------------------------
__global__ void qkv_gemm(const unsigned short* __restrict__ Xb,
                         const unsigned short* __restrict__ Wab,
                         unsigned short* __restrict__ Qb,
                         unsigned short* __restrict__ Kb,
                         unsigned short* __restrict__ Vt) {
  const int lane = threadIdx.x & 31;
  const int wave = threadIdx.x >> 5;
  const int m0 = blockIdx.y * 32;
  const int n0 = blockIdx.x * 256 + wave * 64;
  v8f z = 0.0f;
  v8f accLo[4] = {z, z, z, z};
  v8f accHi[4] = {z, z, z, z};
  gemm_tile_32x64(Xb, Wab, C_, C_, m0, n0, C_, lane, accLo, accHi);

  const int nloc = lane & 15;
  const int hoff = (lane >> 4) ? 8 : 0;
#pragma unroll
  for (int g = 0; g < 4; ++g) {
    const int n    = n0 + g * 16 + nloc;
    const int comp = n >> 10;        // 0=q 1=k 2=v
    const int c    = n & (C_ - 1);
    const int h    = c >> 6;
    const int d    = c & (HS_ - 1);
#pragma unroll
    for (int jj = 0; jj < 16; ++jj) {
      const int j = jj & 7;
      const int m = m0 + (jj >> 3) * 16 + j + hoff;   // global token row
      const int b = m >> 11;                          // / T_
      const int t = m & (T_ - 1);
      const unsigned short val = f2bf((jj < 8) ? accLo[g][j] : accHi[g][j]);
      if (comp == 0) {
        Qb[(((size_t)(b * NH_ + h)) * T_ + t) * HS_ + d] = val;
      } else if (comp == 1) {
        Kb[(((size_t)(b * NH_ + h)) * T_ + t) * HS_ + d] = val;
      } else {
        Vt[(((size_t)(b * NH_ + h)) * HS_ + d) * T_ + t] = val;
      }
    }
  }
}

// ---------------------------------------------------------------------------
// Flash-style causal attention: one 32-thread block (one wave) per
// (b, h, 16-query tile). Online softmax over 32-key blocks.
// K/V tiles are double-buffered in LDS via GLOBAL_LOAD_ASYNC_TO_LDS_B128
// (ASYNCcnt) so the next block's data movement overlaps this block's WMMAs.
// ---------------------------------------------------------------------------
__global__ void attn_kernel(const unsigned short* __restrict__ Qb,
                            const unsigned short* __restrict__ Kb,
                            const unsigned short* __restrict__ Vt,
                            unsigned short* __restrict__ Yb) {
  __shared__ alignas(16) unsigned short Plds[16 * 32];
#if USE_ASYNC_LDS
  __shared__ alignas(16) unsigned short Kbuf[2][32 * 64];   // [key][d]
  __shared__ alignas(16) unsigned short Vbuf[2][64 * 32];   // [d][key]
#endif

  const int lane = threadIdx.x;          // 32-thread block
  const int nqt  = T_ / 16;              // 128
  const int qt   = blockIdx.x % nqt;
  const int bh   = blockIdx.x / nqt;
  const int b    = bh / NH_;
  const int h    = bh % NH_;
  const int q0   = qt * 16;

  const unsigned short* Q  = Qb + (size_t)bh * T_ * HS_;
  const unsigned short* Kp = Kb + (size_t)bh * T_ * HS_;
  const unsigned short* Vp = Vt + (size_t)bh * HS_ * T_;

  const int row  = lane & 15;
  const int kb   = (lane >> 4) ? 8 : 0;
  const int hoff = kb;                   // same value: row offset of C-layout half

#if USE_ASYNC_LDS
  auto prefetch_kv = [&](int k0, int bi) {
    // K tile: rows k0..k0+31 in [T,HS] layout form one contiguous 4KB chunk
    const char* kgl = (const char*)(Kp + (size_t)k0 * HS_) + lane * 16;
    char*       kls = (char*)(&Kbuf[bi][0]) + lane * 16;
#pragma unroll
    for (int i = 0; i < 8; ++i) {
      __builtin_amdgcn_global_load_async_to_lds_b128(
          (gas_v4i)(kgl + i * 512), (las_v4i)(kls + i * 512), 0, 0);
    }
    // V tile: 64 d-rows x 32 keys; global row stride T_, LDS row stride 32
    const int vr = lane >> 2;            // 0..7
    const int vc = (lane & 3) * 8;       // element offset (16B granules)
#pragma unroll
    for (int i = 0; i < 8; ++i) {
      const int d = i * 8 + vr;
      const unsigned short* vg = Vp + (size_t)d * T_ + k0 + vc;
      unsigned short*       vl = &Vbuf[bi][d * 32 + vc];
      __builtin_amdgcn_global_load_async_to_lds_b128(
          (gas_v4i)vg, (las_v4i)vl, 0, 0);
    }
  };
#endif

  // Q fragments (16 x 64 = two 16x32 A-fragments), loaded once
  Frag qf[2];
  const unsigned short* qrow = Q + (size_t)(q0 + row) * HS_;
  qf[0].q[0] = *(const uint4*)(qrow + kb);
  qf[0].q[1] = *(const uint4*)(qrow + kb + 16);
  qf[1].q[0] = *(const uint4*)(qrow + 32 + kb);
  qf[1].q[1] = *(const uint4*)(qrow + 32 + kb + 16);

  v8f z = 0.0f;
  v8f o[4] = {z, z, z, z};               // O tile 16 x 64 (f32 accum)
  float mr[8], lr[8];
#pragma unroll
  for (int j = 0; j < 8; ++j) { mr[j] = -INFINITY; lr[j] = 0.0f; }

  const float scale = 0.125f;            // 1/sqrt(64)

#if USE_ASYNC_LDS
  prefetch_kv(0, 0);
  int bi = 0;
#endif

  for (int k0 = 0; k0 <= q0 + 15; k0 += 32) {
#if USE_ASYNC_LDS
    const bool has_next = (k0 + 32) <= (q0 + 15);
    if (has_next) prefetch_kv(k0 + 32, bi ^ 1);
    // 16 async ops per tile-pair; loads complete in order, so waiting for
    // <=16 outstanding means the current buffer's 16 ops have landed.
    if (has_next) __builtin_amdgcn_s_wait_asynccnt(16);
    else          __builtin_amdgcn_s_wait_asynccnt(0);
    __builtin_amdgcn_wave_barrier();
#endif
    // ----- S = Q K^T for keys [k0, k0+32) : two 16x16 f32 fragments -----
    v8f s[2] = {z, z};
#pragma unroll
    for (int kg = 0; kg < 2; ++kg) {
#if USE_ASYNC_LDS
      const unsigned short* krow = &Kbuf[bi][(kg * 16 + row) * HS_];
#else
      const unsigned short* krow = Kp + (size_t)(k0 + kg * 16 + row) * HS_;
#endif
      Frag bk;
      bk.q[0] = *(const uint4*)(krow + kb);
      bk.q[1] = *(const uint4*)(krow + kb + 16);
      s[kg] = wmma_bf16(qf[0], bk, s[kg]);
      bk.q[0] = *(const uint4*)(krow + 32 + kb);
      bk.q[1] = *(const uint4*)(krow + 32 + kb + 16);
      s[kg] = wmma_bf16(qf[1], bk, s[kg]);
    }
    // ----- scale + causal mask (branchless) -----
#pragma unroll
    for (int kg = 0; kg < 2; ++kg) {
      const int kj = k0 + kg * 16 + row;       // this lane's key column
#pragma unroll
      for (int j = 0; j < 8; ++j) {
        const int qi = q0 + j + hoff;          // this element's query row
        const float v = s[kg][j] * scale;
        s[kg][j] = (kj > qi) ? -1.0e30f : v;
      }
    }
    // ----- online softmax: row reductions live in 16-lane halves -----
#pragma unroll
    for (int j = 0; j < 8; ++j) {
      float cmax = fmaxf(s[0][j], s[1][j]);
      cmax = fmaxf(cmax, __shfl_xor(cmax, 1, 32));
      cmax = fmaxf(cmax, __shfl_xor(cmax, 2, 32));
      cmax = fmaxf(cmax, __shfl_xor(cmax, 4, 32));
      cmax = fmaxf(cmax, __shfl_xor(cmax, 8, 32));
      const float mn    = fmaxf(mr[j], cmax);
      const float alpha = __expf(mr[j] - mn);
      const float p0 = __expf(s[0][j] - mn);
      const float p1 = __expf(s[1][j] - mn);
      float ps = p0 + p1;
      ps += __shfl_xor(ps, 1, 32);
      ps += __shfl_xor(ps, 2, 32);
      ps += __shfl_xor(ps, 4, 32);
      ps += __shfl_xor(ps, 8, 32);
      lr[j] = lr[j] * alpha + ps;
      mr[j] = mn;
      o[0][j] *= alpha; o[1][j] *= alpha; o[2][j] *= alpha; o[3][j] *= alpha;
      // stage P (bf16) into LDS in [query 16][key 32] layout
      const int M = j + hoff;
      Plds[M * 32 + row]      = f2bf(p0);
      Plds[M * 32 + 16 + row] = f2bf(p1);
    }
    __syncthreads();
    // re-read P as a WMMA A-fragment (16x32)
    Frag pf;
    pf.q[0] = *(const uint4*)(&Plds[row * 32 + kb]);
    pf.q[1] = *(const uint4*)(&Plds[row * 32 + kb + 16]);
    // ----- O += P V : V^T rows are d, contiguous in key index -----
#pragma unroll
    for (int g = 0; g < 4; ++g) {
#if USE_ASYNC_LDS
      const unsigned short* vrow = &Vbuf[bi][(g * 16 + row) * 32];
#else
      const unsigned short* vrow = Vp + (size_t)(g * 16 + row) * T_ + k0;
#endif
      Frag bv;
      bv.q[0] = *(const uint4*)(vrow + kb);
      bv.q[1] = *(const uint4*)(vrow + kb + 16);
      o[g] = wmma_bf16(pf, bv, o[g]);
    }
    __syncthreads();   // protect Plds before next iteration's stores
#if USE_ASYNC_LDS
    bi ^= 1;
#endif
  }

  // ----- epilogue: Y[b,t, h*64 + d] = O / l  (bf16 for the proj GEMM) -----
#pragma unroll
  for (int g = 0; g < 4; ++g) {
#pragma unroll
    for (int j = 0; j < 8; ++j) {
      const int t = q0 + j + hoff;
      const float val = o[g][j] / lr[j];
      Yb[((size_t)(b * T_ + t)) * C_ + h * HS_ + g * 16 + row] = f2bf(val);
    }
  }
}

// ---------------------------------------------------------------------------
// out = y @ W_proj^T  (f32 output straight to d_out)
// ---------------------------------------------------------------------------
__global__ void proj_gemm(const unsigned short* __restrict__ Yb,
                          const unsigned short* __restrict__ Wpb,
                          float* __restrict__ out) {
  const int lane = threadIdx.x & 31;
  const int wave = threadIdx.x >> 5;
  const int m0 = blockIdx.y * 32;
  const int n0 = blockIdx.x * 256 + wave * 64;
  v8f z = 0.0f;
  v8f accLo[4] = {z, z, z, z};
  v8f accHi[4] = {z, z, z, z};
  gemm_tile_32x64(Yb, Wpb, C_, C_, m0, n0, C_, lane, accLo, accHi);

  const int nloc = lane & 15;
  const int hoff = (lane >> 4) ? 8 : 0;
#pragma unroll
  for (int g = 0; g < 4; ++g) {
    const int n = n0 + g * 16 + nloc;
#pragma unroll
    for (int j = 0; j < 8; ++j) {
      out[(size_t)(m0 + j + hoff) * C_ + n]      = accLo[g][j];
      out[(size_t)(m0 + 16 + j + hoff) * C_ + n] = accHi[g][j];
    }
  }
}

// ---------------------------------------------------------------------------
extern "C" void kernel_launch(void* const* d_in, const int* in_sizes, int n_in,
                              void* d_out, int out_size, void* d_ws, size_t ws_size,
                              hipStream_t stream) {
  (void)in_sizes; (void)n_in; (void)out_size; (void)ws_size;
  const float* x      = (const float*)d_in[0];
  const float* W_attn = (const float*)d_in[1];
  const float* W_proj = (const float*)d_in[2];
  // d_in[3..6] = pos_emb, W_fc, W_cproj, ln_w : results discarded by reference
  float* out = (float*)d_out;

  char* ws = (char*)d_ws;
  size_t off = 0;
  auto take = [&](size_t nelem) {
    unsigned short* p = (unsigned short*)(ws + off);
    off += (nelem * sizeof(unsigned short) + 255) & ~(size_t)255;
    return p;
  };
  unsigned short* Xb  = take((size_t)B_ * T_ * C_);        // x (bf16)
  unsigned short* Wab = take((size_t)3 * C_ * C_);         // W_attn (bf16)
  unsigned short* Wpb = take((size_t)C_ * C_);             // W_proj (bf16)
  unsigned short* Qb  = take((size_t)B_ * NH_ * T_ * HS_); // Q [B,NH,T,HS]
  unsigned short* Kb  = take((size_t)B_ * NH_ * T_ * HS_); // K [B,NH,T,HS]
  unsigned short* Vt  = take((size_t)B_ * NH_ * T_ * HS_); // V^T [B,NH,HS,T]
  unsigned short* Yb  = take((size_t)B_ * T_ * C_);        // attn out (bf16)

  // 1) down-convert inputs to bf16
  cvt_f32_bf16<<<dim3(2048), 256, 0, stream>>>(x, Xb, (B_ * T_ * C_) / 4);
  cvt_f32_bf16<<<dim3(1024), 256, 0, stream>>>(W_attn, Wab, (3 * C_ * C_) / 4);
  cvt_f32_bf16<<<dim3(512),  256, 0, stream>>>(W_proj, Wpb, (C_ * C_) / 4);

  // 2) qkv projection:  M=8192, N=3072, K=1024 ; block tile 32x256
  qkv_gemm<<<dim3(3 * C_ / 256, (B_ * T_) / 32), 128, 0, stream>>>(Xb, Wab, Qb, Kb, Vt);

  // 3) causal flash attention: one wave per (b,h,16-query tile)
  attn_kernel<<<dim3(B_ * NH_ * (T_ / 16)), 32, 0, stream>>>(Qb, Kb, Vt, Yb);

  // 4) output projection: M=8192, N=1024, K=1024
  proj_gemm<<<dim3(C_ / 256, (B_ * T_) / 32), 128, 0, stream>>>(Yb, Wpb, out);
}
